// AttentionBlock_27419071218280
// MI455X (gfx1250) — compile-verified, matrix operand order checked by reading, not run on hardware
//
#include <hip/hip_runtime.h>
#include <math.h>

// Problem constants (B,H,W,C = 2,64,64,512; GROUPS=32)
#define S_LEN  4096                 // H*W
#define CDIM   512
#define BATCH  2
#define MTOT   (BATCH * S_LEN)      // 8192 rows for batched row-major GEMMs
#define GROUPS 32
#define CPG    (CDIM / GROUPS)      // 16 channels per group
#define GN_N   (S_LEN * CPG)        // 65536 elements per (batch, group)

// Block tile 128x128 = 8 waves * (64x32 wave tile), wave grid 2 (M) x 4 (N)
#define BLK_M  128
#define BLK_N  128
#define WAVE_MT 4                   // 4 x 16 rows per wave
#define WAVE_NT 2                   // 2 x 16 cols per wave

typedef __attribute__((ext_vector_type(16))) __bf16 v16bf;
typedef __attribute__((ext_vector_type(8)))  float  v8f;

union Frag16 { v16bf v; uint4 u[2]; };

__device__ __forceinline__ v8f vzero8() {
  v8f z = {0.f, 0.f, 0.f, 0.f, 0.f, 0.f, 0.f, 0.f};
  return z;
}

// Fragment loader for 16-bit 16x32 A tiles (and, by symmetry, B tiles loaded
// from B^T). Per CDNA5 ISA 16-bit A layout: lanes 0-15 hold rows 0-15 with
// K=0..7 (VGPR0-3) and K=16..23 (VGPR4-7); lanes 16-31 hold the same rows with
// K=8..15 and K=24..31.
__device__ __forceinline__ v16bf load_frag(const __bf16* p, int ld) {
  const int lane = threadIdx.x & 31;
  const int r    = lane & 15;
  const int h    = lane >> 4;
  const __bf16* q = p + (size_t)r * ld + (h << 3);
  Frag16 f;
  f.u[0] = *reinterpret_cast<const uint4*>(q);        // K = base .. base+7
  f.u[1] = *reinterpret_cast<const uint4*>(q + 16);   // K = base+16 .. base+23
  return f.v;
}

__device__ __forceinline__ v8f wmma_bf16(v16bf a, v16bf b, v8f c) {
  // (neg_a, A, neg_b, B, c_mod, C, reuse_a, reuse_b)
  return __builtin_amdgcn_wmma_f32_16x16x32_bf16(false, a, false, b, (short)0,
                                                 c, false, false);
}

// Wave-level GEMM tile: acc[MT][NT] += A[m0:,k] * B[k, n0:]  (B given as B^T)
template<int MT, int NT>
__device__ __forceinline__ void gemm_acc(const __bf16* A, int lda,
                                         const __bf16* BT, int ldb,
                                         int m0, int n0, int K,
                                         v8f acc[MT][NT]) {
  for (int k = 0; k < K; k += 32) {
    v16bf a[MT], b[NT];
#pragma unroll
    for (int i = 0; i < MT; ++i)
      a[i] = load_frag(A + (size_t)(m0 + 16 * i) * lda + k, lda);
#pragma unroll
    for (int j = 0; j < NT; ++j)
      b[j] = load_frag(BT + (size_t)(n0 + 16 * j) * ldb + k, ldb);
#pragma unroll
    for (int i = 0; i < MT; ++i)
#pragma unroll
      for (int j = 0; j < NT; ++j)
        acc[i][j] = wmma_bf16(a[i], b[j], acc[i][j]);
  }
}

// Wave/lane coordinates shared by all GEMM kernels.
struct Tile {
  int m0, n0;    // wave tile origin
  int r, h;      // lane%16 (column within 16-tile), lane/16 (row-half select)
};
__device__ __forceinline__ Tile tile_coords() {
  Tile t;
  const int wave = threadIdx.x >> 5;      // 8 waves
  const int wm = wave & 1;                // 2 waves in M
  const int wn = wave >> 1;               // 4 waves in N
  t.m0 = blockIdx.x * BLK_M + wm * (16 * WAVE_MT);
  t.n0 = blockIdx.y * BLK_N + wn * (16 * WAVE_NT);
  const int lane = threadIdx.x & 31;
  t.r = lane & 15;
  t.h = lane >> 4;
  return t;
}

// ---------------------------------------------------------------------------
// 0) Weight transpose + fp32 -> bf16 cast: WT[n,k] = W[k,n]
// ---------------------------------------------------------------------------
__global__ void k_transpose_cast(const float* __restrict__ W,
                                 __bf16* __restrict__ WT) {
  int i = blockIdx.x * 256 + threadIdx.x;        // 512*512 elements
  int k = i >> 9;
  int n = i & 511;
  WT[(size_t)n * CDIM + k] = (__bf16)W[i];
}

// ---------------------------------------------------------------------------
// 1) GroupNorm over (h,w,c/G) per (batch, group); emit normalized x as bf16
// ---------------------------------------------------------------------------
__global__ void k_groupnorm(const float* __restrict__ x,
                            const float* __restrict__ gamma,
                            const float* __restrict__ beta,
                            __bf16* __restrict__ xn) {
  const int b = blockIdx.x >> 5;
  const int g = blockIdx.x & 31;
  const int tid = threadIdx.x;                   // 256 threads
  const float* base = x + (size_t)b * S_LEN * CDIM + g * CPG;

  float s = 0.f, ss = 0.f;
  for (int p = tid; p < S_LEN; p += 256) {
    const float4* q = reinterpret_cast<const float4*>(base + (size_t)p * CDIM);
#pragma unroll
    for (int t = 0; t < 4; ++t) {
      float4 v = q[t];
      s  += v.x + v.y + v.z + v.w;
      ss += v.x * v.x + v.y * v.y + v.z * v.z + v.w * v.w;
    }
  }
  __shared__ float r0[256], r1[256];
  r0[tid] = s; r1[tid] = ss;
  __syncthreads();
  for (int o = 128; o; o >>= 1) {
    if (tid < o) { r0[tid] += r0[tid + o]; r1[tid] += r1[tid + o]; }
    __syncthreads();
  }
  const float inv_n = 1.f / (float)GN_N;
  const float mu   = r0[0] * inv_n;
  const float var  = r1[0] * inv_n - mu * mu;
  const float rstd = rsqrtf(var + 1e-5f);

  float gg[CPG], bb[CPG];
#pragma unroll
  for (int t = 0; t < CPG; ++t) { gg[t] = gamma[g * CPG + t]; bb[t] = beta[g * CPG + t]; }

  __bf16* ob = xn + (size_t)b * S_LEN * CDIM + g * CPG;
  for (int p = tid; p < S_LEN; p += 256) {
    const float* src = base + (size_t)p * CDIM;
    __bf16*      dst = ob   + (size_t)p * CDIM;
#pragma unroll
    for (int t = 0; t < CPG; ++t)
      dst[t] = (__bf16)((src[t] - mu) * rstd * gg[t] + bb[t]);
  }
}

// ---------------------------------------------------------------------------
// 2) Projection GEMM: out = xn @ W + b.
//    VT_MODE=0 -> bf16 [M,512] row-major; VT_MODE=1 -> per-batch V^T [b][c][s]
// ---------------------------------------------------------------------------
template<int VT_MODE>
__global__ void k_gemm_qkv(const __bf16* __restrict__ A,
                           const __bf16* __restrict__ WT,
                           const float* __restrict__ bias,
                           __bf16* __restrict__ out) {
  const Tile t = tile_coords();

  v8f acc[WAVE_MT][WAVE_NT];
#pragma unroll
  for (int i = 0; i < WAVE_MT; ++i)
#pragma unroll
    for (int j = 0; j < WAVE_NT; ++j) acc[i][j] = vzero8();

  gemm_acc<WAVE_MT, WAVE_NT>(A, CDIM, WT, CDIM, t.m0, t.n0, CDIM, acc);

#pragma unroll
  for (int i = 0; i < WAVE_MT; ++i)
#pragma unroll
    for (int j = 0; j < WAVE_NT; ++j) {
      const int col = t.n0 + 16 * j + t.r;
      const float bv = bias[col];
#pragma unroll
      for (int e = 0; e < 8; ++e) {
        const int row = t.m0 + 16 * i + e + 8 * t.h;
        const float v = acc[i][j][e] + bv;
        if (VT_MODE == 0) {
          out[(size_t)row * CDIM + col] = (__bf16)v;
        } else {
          const int b = row >> 12;       // row / 4096
          const int sIdx = row & 4095;
          out[((size_t)b * CDIM + col) * S_LEN + sIdx] = (__bf16)v;
        }
      }
    }
}

// ---------------------------------------------------------------------------
// 3) Raw attention scores: S[b,q,k] = scale * <Q_q, K_k>   (bf16 out)
// ---------------------------------------------------------------------------
__global__ void k_gemm_scores(const __bf16* __restrict__ Q,
                              const __bf16* __restrict__ Km,
                              __bf16* __restrict__ S, float scale) {
  const int b = blockIdx.z;
  const __bf16* A  = Q  + (size_t)b * S_LEN * CDIM;
  const __bf16* BT = Km + (size_t)b * S_LEN * CDIM;  // rows of K == cols of K^T
  __bf16* out = S + (size_t)b * S_LEN * S_LEN;

  const Tile t = tile_coords();

  v8f acc[WAVE_MT][WAVE_NT];
#pragma unroll
  for (int i = 0; i < WAVE_MT; ++i)
#pragma unroll
    for (int j = 0; j < WAVE_NT; ++j) acc[i][j] = vzero8();

  gemm_acc<WAVE_MT, WAVE_NT>(A, CDIM, BT, CDIM, t.m0, t.n0, CDIM, acc);

#pragma unroll
  for (int i = 0; i < WAVE_MT; ++i)
#pragma unroll
    for (int j = 0; j < WAVE_NT; ++j) {
      const int col = t.n0 + 16 * j + t.r;
#pragma unroll
      for (int e = 0; e < 8; ++e) {
        const int row = t.m0 + 16 * i + e + 8 * t.h;
        out[(size_t)row * S_LEN + col] = (__bf16)(acc[i][j][e] * scale);
      }
    }
}

// ---------------------------------------------------------------------------
// 4) Row-wise online softmax stats (max, sumexp). One wave per row.
// ---------------------------------------------------------------------------
__global__ void k_row_stats(const __bf16* __restrict__ S,
                            float* __restrict__ Mr, float* __restrict__ Lr) {
  const int row  = blockIdx.x * 8 + (threadIdx.x >> 5);  // 8 waves/block
  const int lane = threadIdx.x & 31;
  const __bf16* p = S + (size_t)row * S_LEN;

  float m = -INFINITY, l = 0.f;
  for (int c = lane; c < S_LEN; c += 32) {
    const float v = (float)p[c];
    const float nm = fmaxf(m, v);
    l = l * __expf(m - nm) + __expf(v - nm);
    m = nm;
  }
  for (int off = 16; off; off >>= 1) {
    const float mo = __shfl_xor(m, off, 32);
    const float lo = __shfl_xor(l, off, 32);
    const float nm = fmaxf(m, mo);
    l = l * __expf(m - nm) + lo * __expf(mo - nm);
    m = nm;
  }
  if (lane == 0) { Mr[row] = m; Lr[row] = l; }
}

// ---------------------------------------------------------------------------
// 5) Normalize scores in place: P = exp(s - m) / l
// ---------------------------------------------------------------------------
__global__ void k_softmax_norm(__bf16* __restrict__ S,
                               const float* __restrict__ Mr,
                               const float* __restrict__ Lr) {
  const size_t i = (size_t)blockIdx.x * 256 + threadIdx.x;
  const int row = (int)(i >> 12);            // 4096 cols
  const float v = (float)S[i];
  S[i] = (__bf16)(__expf(v - Mr[row]) / Lr[row]);
}

// ---------------------------------------------------------------------------
// 6) O = P @ V    (A = P [4096x4096], B^T = V^T [512 x 4096]) -> bf16 [s,c]
// ---------------------------------------------------------------------------
__global__ void k_gemm_pv(const __bf16* __restrict__ P,
                          const __bf16* __restrict__ VT,
                          __bf16* __restrict__ O) {
  const int b = blockIdx.z;
  const __bf16* A  = P  + (size_t)b * S_LEN * S_LEN;
  const __bf16* BT = VT + (size_t)b * CDIM * S_LEN;
  __bf16* out = O + (size_t)b * S_LEN * CDIM;

  const Tile t = tile_coords();

  v8f acc[WAVE_MT][WAVE_NT];
#pragma unroll
  for (int i = 0; i < WAVE_MT; ++i)
#pragma unroll
    for (int j = 0; j < WAVE_NT; ++j) acc[i][j] = vzero8();

  gemm_acc<WAVE_MT, WAVE_NT>(A, S_LEN, BT, S_LEN, t.m0, t.n0, S_LEN, acc);

#pragma unroll
  for (int i = 0; i < WAVE_MT; ++i)
#pragma unroll
    for (int j = 0; j < WAVE_NT; ++j) {
      const int col = t.n0 + 16 * j + t.r;
#pragma unroll
      for (int e = 0; e < 8; ++e) {
        const int row = t.m0 + 16 * i + e + 8 * t.h;
        out[(size_t)row * CDIM + col] = (__bf16)acc[i][j][e];
      }
    }
}

// ---------------------------------------------------------------------------
// 7) Final: out = O @ Wp + bp + residual   (fp32 out)
// ---------------------------------------------------------------------------
__global__ void k_gemm_out(const __bf16* __restrict__ O,
                           const __bf16* __restrict__ WpT,
                           const float* __restrict__ bp,
                           const float* __restrict__ resid,
                           float* __restrict__ out) {
  const Tile t = tile_coords();

  v8f acc[WAVE_MT][WAVE_NT];
#pragma unroll
  for (int i = 0; i < WAVE_MT; ++i)
#pragma unroll
    for (int j = 0; j < WAVE_NT; ++j) acc[i][j] = vzero8();

  gemm_acc<WAVE_MT, WAVE_NT>(O, CDIM, WpT, CDIM, t.m0, t.n0, CDIM, acc);

#pragma unroll
  for (int i = 0; i < WAVE_MT; ++i)
#pragma unroll
    for (int j = 0; j < WAVE_NT; ++j) {
      const int col = t.n0 + 16 * j + t.r;
      const float bv = bp[col];
#pragma unroll
      for (int e = 0; e < 8; ++e) {
        const int row = t.m0 + 16 * i + e + 8 * t.h;
        const size_t idx = (size_t)row * CDIM + col;
        out[idx] = acc[i][j][e] + bv + resid[idx];
      }
    }
}

// ---------------------------------------------------------------------------
extern "C" void kernel_launch(void* const* d_in, const int* in_sizes, int n_in,
                              void* d_out, int out_size, void* d_ws, size_t ws_size,
                              hipStream_t stream) {
  (void)in_sizes; (void)n_in; (void)out_size; (void)ws_size;

  const float* x_in  = (const float*)d_in[0];
  const float* gamma = (const float*)d_in[1];
  const float* beta  = (const float*)d_in[2];
  const float* Wq = (const float*)d_in[3];  const float* bq = (const float*)d_in[4];
  const float* Wk = (const float*)d_in[5];  const float* bk = (const float*)d_in[6];
  const float* Wv = (const float*)d_in[7];  const float* bv = (const float*)d_in[8];
  const float* Wp = (const float*)d_in[9];  const float* bp = (const float*)d_in[10];
  float* out = (float*)d_out;

  // Workspace carve-up (all offsets 256B aligned)
  char* ws = (char*)d_ws;
  auto carve = [&](size_t bytes) {
    char* p = ws;
    ws += (bytes + 255) & ~(size_t)255;
    return p;
  };
  __bf16* xn  = (__bf16*)carve((size_t)MTOT * CDIM * 2);          // 8 MB
  __bf16* WqT = (__bf16*)carve((size_t)CDIM * CDIM * 2);
  __bf16* WkT = (__bf16*)carve((size_t)CDIM * CDIM * 2);
  __bf16* WvT = (__bf16*)carve((size_t)CDIM * CDIM * 2);
  __bf16* WpT = (__bf16*)carve((size_t)CDIM * CDIM * 2);
  __bf16* Qb  = (__bf16*)carve((size_t)MTOT * CDIM * 2);          // 8 MB
  __bf16* Kb  = (__bf16*)carve((size_t)MTOT * CDIM * 2);          // 8 MB
  __bf16* VTb = (__bf16*)carve((size_t)MTOT * CDIM * 2);          // 8 MB (per-batch [C][S])
  __bf16* Ob  = (__bf16*)carve((size_t)MTOT * CDIM * 2);          // 8 MB
  float*  Mr  = (float*)carve((size_t)MTOT * 4);
  float*  Lr  = (float*)carve((size_t)MTOT * 4);
  __bf16* Pb  = (__bf16*)carve((size_t)BATCH * S_LEN * S_LEN * 2);// 67 MB (L2-resident)

  const dim3 blk(256);

  // 0) weights -> bf16 transposed
  k_transpose_cast<<<dim3(CDIM * CDIM / 256), blk, 0, stream>>>(Wq, WqT);
  k_transpose_cast<<<dim3(CDIM * CDIM / 256), blk, 0, stream>>>(Wk, WkT);
  k_transpose_cast<<<dim3(CDIM * CDIM / 256), blk, 0, stream>>>(Wv, WvT);
  k_transpose_cast<<<dim3(CDIM * CDIM / 256), blk, 0, stream>>>(Wp, WpT);

  // 1) group norm -> xn (bf16)
  k_groupnorm<<<dim3(BATCH * GROUPS), blk, 0, stream>>>(x_in, gamma, beta, xn);

  // 2) Q, K row-major; V stored transposed [b][c][s]
  const dim3 gProj(MTOT / BLK_M, CDIM / BLK_N);
  k_gemm_qkv<0><<<gProj, blk, 0, stream>>>(xn, WqT, bq, Qb);
  k_gemm_qkv<0><<<gProj, blk, 0, stream>>>(xn, WkT, bk, Kb);
  k_gemm_qkv<1><<<gProj, blk, 0, stream>>>(xn, WvT, bv, VTb);

  // 3) scores S = scale * Q K^T
  const float scale = 1.0f / sqrtf((float)CDIM);
  k_gemm_scores<<<dim3(S_LEN / BLK_M, S_LEN / BLK_N, BATCH), blk, 0, stream>>>(Qb, Kb, Pb, scale);

  // 4) row stats (max, sumexp)
  k_row_stats<<<dim3(MTOT / 8), blk, 0, stream>>>(Pb, Mr, Lr);

  // 5) normalize in place -> P
  k_softmax_norm<<<dim3((unsigned)((size_t)MTOT * S_LEN / 256)), blk, 0, stream>>>(Pb, Mr, Lr);

  // 6) O = P @ V
  k_gemm_pv<<<dim3(S_LEN / BLK_M, CDIM / BLK_N, BATCH), blk, 0, stream>>>(Pb, VTb, Ob);

  // 7) out = O @ Wp + bp + residual
  k_gemm_out<<<gProj, blk, 0, stream>>>(Ob, WpT, bp, x_in, out);
}
